// DogeCDMoE_56349970923544
// MI455X (gfx1250) — matrix operand order repaired and checked by reading, not compiled, not run on hardware
//
#include <hip/hip_runtime.h>

#define HIDDEN   1024
#define INTER    2048
#define NUM_KEYS 128
#define TOPK     64
#define NTOK     2048

typedef __attribute__((ext_vector_type(16))) __bf16 v16bf;
typedef __attribute__((ext_vector_type(8)))  float  v8f;
typedef __attribute__((ext_vector_type(2)))  float  v2f;

union ABf { v16bf v; unsigned short s[16]; uint4 q[2]; };

__device__ __forceinline__ unsigned short f2bf(float f) {
    unsigned int u = __float_as_uint(f);
    u += 0x7fffu + ((u >> 16) & 1u);          // round-to-nearest-even
    return (unsigned short)(u >> 16);
}

// LDS byte offset of a __shared__ object (generic LDS ptr low 32 bits = AS3 offset)
__device__ __forceinline__ unsigned ldsOff(const void* p) {
    return (unsigned)(unsigned long long)p;
}

// gfx1250 async copy: per-lane 16B global -> LDS, tracked by ASYNCcnt (FIFO)
__device__ __forceinline__ void async_ld_b128(unsigned lds_off, const void* gaddr) {
    asm volatile("global_load_async_to_lds_b128 %0, %1, off"
                 :: "v"(lds_off), "v"(gaddr) : "memory");
}
#define WAIT_ASYNC(n) asm volatile("s_wait_asynccnt " #n ::: "memory")

// ---------------------------------------------------------------- convert
__global__ void cvt_f32_bf16(const float* __restrict__ in,
                             unsigned short* __restrict__ out, int n) {
    int i = blockIdx.x * blockDim.x + threadIdx.x;
    int stride = gridDim.x * blockDim.x;
    for (; i < n; i += stride) out[i] = f2bf(in[i]);
}

// ---------------------------------------------------------- router (f32 WMMA)
// R (flat [N,256] row-major) = X[2048,1024] @ Wr[256,1024]^T
__global__ void __launch_bounds__(256)
router_gemm(const float* __restrict__ X, const float* __restrict__ W,
            float* __restrict__ R) {
    int wave = (blockIdx.x * blockDim.x + threadIdx.x) >> 5;
    int lane = threadIdx.x & 31;
    int nt = wave & 15;               // 256/16 col tiles
    int mt = wave >> 4;               // 2048/16 row tiles
    int hi = lane >> 4;               // lane half selects K pair
    int row = mt * 16 + (lane & 15);  // A: M = lane%16 in both halves
    int col = nt * 16 + (lane & 15);  // B/D: N = lane%16
    const float* ar = X + (long)row * HIDDEN;
    const float* br = W + (long)col * HIDDEN;   // B[k][n] = Wr[n][k]
    v8f acc = {};
    for (int k = 0; k < HIDDEN; k += 4) {
        int ko = k + hi * 2;
        v2f a, b;
        a.x = ar[ko]; a.y = ar[ko + 1];
        b.x = br[ko]; b.y = br[ko + 1];
        acc = __builtin_amdgcn_wmma_f32_16x16x4_f32(
            false, a, false, b, (short)0, acc, false, false);
    }
    int rbase = mt * 16 + hi * 8;     // D: M = v + 8*laneHi
    for (int v = 0; v < 8; ++v)
        R[(long)(rbase + v) * (2 * NUM_KEYS) + col] = acc[v];
}

// ------------------------------------------------------------- routing topk
__global__ void __launch_bounds__(256)
route_topk(const float* __restrict__ R, int* __restrict__ topIdx,
           float* __restrict__ topW) {
    __shared__ float sv[2][128];
    __shared__ int   si[2][128];
    __shared__ float candv[4096];
    __shared__ int   candi[4096];
    __shared__ float redv[256];
    __shared__ int   redp[256];
    __shared__ float topv[TOPK];
    __shared__ int   topi[TOPK];
    __shared__ float esum;

    int t = blockIdx.x, tid = threadIdx.x;
    int arr = tid >> 7, e = tid & 127;
    // reshape(2,N,128) reinterpretation of flat [N,256]
    sv[arr][e] = R[((long)arr * NTOK + t) * NUM_KEYS + e];
    si[arr][e] = e;
    __syncthreads();

    // bitonic sort both lists descending (value,index), 128 threads each
    for (int size = 2; size <= 128; size <<= 1) {
        for (int j = size >> 1; j > 0; j >>= 1) {
            int l = e ^ j;
            if (l > e) {
                bool up = ((e & size) == 0);
                float ve = sv[arr][e], vl = sv[arr][l];
                if ((ve < vl) == up) {
                    sv[arr][e] = vl; sv[arr][l] = ve;
                    int ie = si[arr][e]; si[arr][e] = si[arr][l]; si[arr][l] = ie;
                }
            }
            __syncthreads();
        }
    }
    // 64x64 candidate sums (superset of global top-64 by dominance)
    for (int c = tid; c < 4096; c += 256) {
        int i = c >> 6, j = c & 63;
        candv[c] = sv[0][i] + sv[1][j];
        candi[c] = (si[0][i] << 7) | si[1][j];
    }
    __syncthreads();
    // 64 rounds of block-wide argmax
    for (int k = 0; k < TOPK; ++k) {
        float m = -3.4e38f; int mp = 0;
        for (int c = tid; c < 4096; c += 256) {
            float v = candv[c];
            if (v > m) { m = v; mp = c; }
        }
        redv[tid] = m; redp[tid] = mp;
        __syncthreads();
        for (int s = 128; s > 0; s >>= 1) {
            if (tid < s && redv[tid + s] > redv[tid]) {
                redv[tid] = redv[tid + s]; redp[tid] = redp[tid + s];
            }
            __syncthreads();
        }
        if (tid == 0) {
            int p = redp[0];
            topv[k] = redv[0]; topi[k] = candi[p];
            candv[p] = -3.4e38f;
        }
        __syncthreads();
    }
    // softmax (topv[0] is the max)
    if (tid < TOPK) redv[tid] = __expf(topv[tid] - topv[0]);
    __syncthreads();
    if (tid == 0) {
        float s = 0.f;
        for (int i = 0; i < TOPK; ++i) s += redv[i];
        esum = s;
    }
    __syncthreads();
    if (tid < TOPK) {
        topW[(long)t * TOPK + tid]   = redv[tid] / esum;
        topIdx[(long)t * TOPK + tid] = topi[tid];
    }
}

// ------------------------------------------------- shared MLP GEMM1 (bf16 WMMA)
// Im[2048,2048] = silu(X@Wg^T) * (X@Wu^T)
// block tile M=128 x N=32; double-buffered async global->LDS staging
__global__ void __launch_bounds__(256)
mlp_gemm1(const unsigned short* __restrict__ Xb,
          const unsigned short* __restrict__ Gb,
          const unsigned short* __restrict__ Ub,
          unsigned short* __restrict__ Im) {
    __shared__ __align__(16) unsigned short lA[2][128 * 64];  // 2 x 16 KB
    __shared__ __align__(16) unsigned short lG[2][32 * 64];   // 2 x  4 KB
    __shared__ __align__(16) unsigned short lU[2][32 * 64];   // 2 x  4 KB

    int tid = threadIdx.x;
    int wv = tid >> 5, lane = tid & 31;
    int nb = blockIdx.x & 63;    // 2048/32 col blocks
    int mb = blockIdx.x >> 6;    // 2048/128 row blocks
    int hi = lane >> 4;
    int ln = lane & 15;

    unsigned aO[2] = { ldsOff(lA[0]), ldsOff(lA[1]) };
    unsigned gO[2] = { ldsOff(lG[0]), ldsOff(lG[1]) };
    unsigned uO[2] = { ldsOff(lU[0]), ldsOff(lU[1]) };

    // per-thread staging source pointers (k0 added per stage)
    const unsigned short* aSrc[4];
#pragma unroll
    for (int r = 0; r < 4; ++r) {
        int c = tid + r * 256;
        aSrc[r] = Xb + (long)(mb * 128 + (c >> 3)) * HIDDEN + (c & 7) * 8;
    }
    const unsigned short* gSrc = Gb + (long)(nb * 32 + (tid >> 3)) * HIDDEN + (tid & 7) * 8;
    const unsigned short* uSrc = Ub + (long)(nb * 32 + (tid >> 3)) * HIDDEN + (tid & 7) * 8;

#define STAGE1(k0, p) do {                                                  \
        _Pragma("unroll")                                                   \
        for (int r = 0; r < 4; ++r)                                         \
            async_ld_b128(aO[p] + (tid + r * 256) * 16, aSrc[r] + (k0));    \
        async_ld_b128(gO[p] + tid * 16, gSrc + (k0));                       \
        async_ld_b128(uO[p] + tid * 16, uSrc + (k0));                       \
    } while (0)

    v8f accg[2] = {}, accu[2] = {};

    STAGE1(0, 0);
    const int NST = HIDDEN / 64;   // 16 stages
    for (int i = 0; i < NST; ++i) {
        int p = i & 1;
        if (i + 1 < NST) {
            STAGE1((i + 1) * 64, p ^ 1);
            WAIT_ASYNC(6);         // stage i's 6 transfers done (FIFO)
        } else {
            WAIT_ASYNC(0);
        }
        __syncthreads();

#pragma unroll
        for (int s = 0; s < 2; ++s) {           // two k=32 substeps
            ABf a;
            int ka = s * 32 + hi * 8;           // A: K 0-7/16-23 or 8-15/24-31
            const unsigned short* ar = lA[p] + (wv * 16 + ln) * 64;
            a.q[0] = *(const uint4*)(ar + ka);
            a.q[1] = *(const uint4*)(ar + ka + 16);
            int kb = s * 32 + hi * 8;           // B: K 0-7/16-23 or 8-15/24-31
#pragma unroll
            for (int nsub = 0; nsub < 2; ++nsub) {
                ABf bg, bu;
                const unsigned short* gr = lG[p] + (nsub * 16 + ln) * 64;
                bg.q[0] = *(const uint4*)(gr + kb);
                bg.q[1] = *(const uint4*)(gr + kb + 16);
                const unsigned short* ur = lU[p] + (nsub * 16 + ln) * 64;
                bu.q[0] = *(const uint4*)(ur + kb);
                bu.q[1] = *(const uint4*)(ur + kb + 16);
                accg[nsub] = __builtin_amdgcn_wmma_f32_16x16x32_bf16(
                    false, a.v, false, bg.v, (short)0, accg[nsub], false, false);
                accu[nsub] = __builtin_amdgcn_wmma_f32_16x16x32_bf16(
                    false, a.v, false, bu.v, (short)0, accu[nsub], false, false);
            }
        }
        __syncthreads();    // buffer p free before it is re-staged at i+1
    }
#undef STAGE1

    int rbase = mb * 128 + wv * 16 + hi * 8;    // D: M = v + 8*laneHi
#pragma unroll
    for (int nsub = 0; nsub < 2; ++nsub) {
        int nc = nb * 32 + nsub * 16 + ln;
#pragma unroll
        for (int v = 0; v < 8; ++v) {
            float gv = accg[nsub][v], uv = accu[nsub][v];
            float h = gv / (1.0f + __expf(-gv)) * uv;   // silu(g)*u
            Im[(long)(rbase + v) * INTER + nc] = f2bf(h);
        }
    }
}

// ------------------------------------------------- shared MLP GEMM2 (bf16 WMMA)
// Out[2048,1024] = Im[2048,2048] @ Wd[1024,2048]^T; block tile M=128 x N=32
__global__ void __launch_bounds__(256)
mlp_gemm2(const unsigned short* __restrict__ Im,
          const unsigned short* __restrict__ Db,
          float* __restrict__ Out) {
    __shared__ __align__(16) unsigned short lA[2][128 * 64];  // 2 x 16 KB
    __shared__ __align__(16) unsigned short lB[2][32 * 64];   // 2 x  4 KB

    int tid = threadIdx.x;
    int wv = tid >> 5, lane = tid & 31;
    int nb = blockIdx.x & 31;    // 1024/32 col blocks
    int mb = blockIdx.x >> 5;    // 2048/128 row blocks
    int hi = lane >> 4;
    int ln = lane & 15;

    unsigned aO[2] = { ldsOff(lA[0]), ldsOff(lA[1]) };
    unsigned bO[2] = { ldsOff(lB[0]), ldsOff(lB[1]) };

    const unsigned short* aSrc[4];
#pragma unroll
    for (int r = 0; r < 4; ++r) {
        int c = tid + r * 256;
        aSrc[r] = Im + (long)(mb * 128 + (c >> 3)) * INTER + (c & 7) * 8;
    }
    const unsigned short* bSrc = Db + (long)(nb * 32 + (tid >> 3)) * INTER + (tid & 7) * 8;

#define STAGE2(k0, p) do {                                                  \
        _Pragma("unroll")                                                   \
        for (int r = 0; r < 4; ++r)                                         \
            async_ld_b128(aO[p] + (tid + r * 256) * 16, aSrc[r] + (k0));    \
        async_ld_b128(bO[p] + tid * 16, bSrc + (k0));                       \
    } while (0)

    v8f acc[2] = {};

    STAGE2(0, 0);
    const int NST = INTER / 64;    // 32 stages
    for (int i = 0; i < NST; ++i) {
        int p = i & 1;
        if (i + 1 < NST) {
            STAGE2((i + 1) * 64, p ^ 1);
            WAIT_ASYNC(5);
        } else {
            WAIT_ASYNC(0);
        }
        __syncthreads();

#pragma unroll
        for (int s = 0; s < 2; ++s) {
            ABf a;
            int ka = s * 32 + hi * 8;
            const unsigned short* ar = lA[p] + (wv * 16 + ln) * 64;
            a.q[0] = *(const uint4*)(ar + ka);
            a.q[1] = *(const uint4*)(ar + ka + 16);
            int kb = s * 32 + hi * 8;
#pragma unroll
            for (int nsub = 0; nsub < 2; ++nsub) {
                ABf b;
                const unsigned short* br = lB[p] + (nsub * 16 + ln) * 64;
                b.q[0] = *(const uint4*)(br + kb);
                b.q[1] = *(const uint4*)(br + kb + 16);
                acc[nsub] = __builtin_amdgcn_wmma_f32_16x16x32_bf16(
                    false, a.v, false, b.v, (short)0, acc[nsub], false, false);
            }
        }
        __syncthreads();
    }
#undef STAGE2

    int rbase = mb * 128 + wv * 16 + hi * 8;
#pragma unroll
    for (int nsub = 0; nsub < 2; ++nsub) {
        int nc = nb * 32 + nsub * 16 + ln;
#pragma unroll
        for (int v = 0; v < 8; ++v)
            Out[(long)(rbase + v) * HIDDEN + nc] = acc[nsub][v];
    }
}

// --------------------------------------------------------- expert gather+combine
__global__ void __launch_bounds__(256)
expert_combine(const float* __restrict__ X,
               const float* __restrict__ Dn,
               const float* __restrict__ Up,
               const int* __restrict__ topIdx,
               const float* __restrict__ topW,
               float* __restrict__ Out) {
    __shared__ __align__(16) float xs[HIDDEN];
    __shared__ float ew[TOPK];
    __shared__ int   eidx[TOPK];
    int t = blockIdx.x, tid = threadIdx.x;
    ((float4*)xs)[tid] = ((const float4*)(X + (long)t * HIDDEN))[tid];
    if (tid < TOPK) eidx[tid] = topIdx[(long)t * TOPK + tid];
    __syncthreads();

    int wv = tid >> 5, lane = tid & 31;
    for (int kk = 0; kk < 8; ++kk) {        // 8 waves x 8 experts
        int k = wv * 8 + kk;
        const float* dr = Dn + (long)eidx[k] * HIDDEN;
        float s = 0.f;
#pragma unroll
        for (int j = 0; j < 8; ++j) {
            int h = lane * 4 + j * 128;
            float4 d  = *(const float4*)(dr + h);
            float4 xv = *(const float4*)(xs + h);
            s += d.x * xv.x + d.y * xv.y + d.z * xv.z + d.w * xv.w;
        }
        for (int m = 16; m > 0; m >>= 1) s += __shfl_xor(s, m, 32);
        if (lane == 0) {
            float g = s / (1.0f + __expf(-s));   // silu
            ew[k] = g * topW[(long)t * TOPK + k];
        }
    }
    __syncthreads();

    float4 acc = {0.f, 0.f, 0.f, 0.f};
#pragma unroll 8
    for (int k = 0; k < TOPK; ++k) {
        const float4* ur = (const float4*)(Up + (long)eidx[k] * HIDDEN);
        float4 u = ur[tid];
        float c = ew[k];
        acc.x += c * u.x; acc.y += c * u.y; acc.z += c * u.z; acc.w += c * u.w;
    }
    float4* orow = (float4*)(Out + (long)t * HIDDEN);
    float4 p = orow[tid];
    p.x += acc.x; p.y += acc.y; p.z += acc.z; p.w += acc.w;
    orow[tid] = p;
}

// -------------------------------------------------------------------- launch
extern "C" void kernel_launch(void* const* d_in, const int* in_sizes, int n_in,
                              void* d_out, int out_size, void* d_ws, size_t ws_size,
                              hipStream_t stream) {
    const float* X  = (const float*)d_in[0];  // [1,2048,1024]
    const float* Wr = (const float*)d_in[1];  // [256,1024]
    const float* Wg = (const float*)d_in[2];  // [2048,1024]
    const float* Wu = (const float*)d_in[3];  // [2048,1024]
    const float* Wd = (const float*)d_in[4];  // [1024,2048]
    const float* De = (const float*)d_in[5];  // [16384,1024]
    const float* Ue = (const float*)d_in[6];  // [16384,1024]

    float* Out  = (float*)d_out;                    // [2048*1024]
    float* Rout = Out + (long)NTOK * HIDDEN;        // router logits [N*256] flat

    unsigned short* Xb = (unsigned short*)d_ws;                 // 4 MB
    unsigned short* Gb = Xb + (long)NTOK * HIDDEN;              // 4 MB
    unsigned short* Ub = Gb + (long)INTER * HIDDEN;             // 4 MB
    unsigned short* Db = Ub + (long)INTER * HIDDEN;             // 4 MB
    unsigned short* Im = Db + (long)HIDDEN * INTER;             // 8 MB
    int*   topIdx = (int*)(Im + (long)NTOK * INTER);            // 512 KB
    float* topW   = (float*)(topIdx + (long)NTOK * TOPK);       // 512 KB

    cvt_f32_bf16<<<256, 256, 0, stream>>>(X,  Xb, NTOK * HIDDEN);
    cvt_f32_bf16<<<256, 256, 0, stream>>>(Wg, Gb, INTER * HIDDEN);
    cvt_f32_bf16<<<256, 256, 0, stream>>>(Wu, Ub, INTER * HIDDEN);
    cvt_f32_bf16<<<256, 256, 0, stream>>>(Wd, Db, HIDDEN * INTER);

    router_gemm<<<(128 * 16) / 8, 256, 0, stream>>>(X, Wr, Rout);
    route_topk<<<NTOK, 256, 0, stream>>>(Rout, topIdx, topW);
    mlp_gemm1<<<16 * 64, 256, 0, stream>>>(Xb, Gb, Ub, Im);
    mlp_gemm2<<<16 * 32, 256, 0, stream>>>(Im, Db, Out);
    expert_combine<<<NTOK, 256, 0, stream>>>(X, De, Ue, topIdx, topW, Out);
}
